// Attention_29076928594416
// MI455X (gfx1250) — matrix operand order repaired and checked by reading, not compiled, not run on hardware
//
#include <hip/hip_runtime.h>
#include <hip/hip_bf16.h>

// ---------------------------------------------------------------------------
// MI455X (gfx1250) relative-position attention, bf16 WMMA pipeline.
// Roofline: ~45 GFLOP vs ~55 MB of HBM traffic -> matrix-core bound; all
// GEMM-like work uses v_wmma_f32_16x16x32_bf16 (wave32, 16x16 tiles, f32 acc).
// Workspace layout (assumes ws_size >= 16 MB):
//   [ 0MB) Qcb bf16 (h,n,dk)   q*scale + content_bias
//   [ 2MB) Qpb bf16 (h,n,dk)   q*scale + pos_bias
//   [ 4MB) K   bf16 (h,n,dk)
//   [ 6MB) Vt  bf16 (h,dv,n)   V transposed for contiguous B-tiles
//   [ 8MB) RelK bf16 (h,8192,dk)  rel-pos keys (row 8191 zeroed pad)
//   [12MB) O   f32  (h,n,dv)   per-head attention output
// ---------------------------------------------------------------------------

#define NSEQ   4096
#define DIMM   1536
#define NHEAD  4
#define DK     64
#define QSCALE 0.125f   // 64^-0.5

typedef __attribute__((ext_vector_type(16))) __bf16 v16bf;
typedef __attribute__((ext_vector_type(8)))  __bf16 v8bf;
typedef __attribute__((ext_vector_type(8)))  float  v8f;

union V16U { v16bf v; v8bf h[2]; };

static __device__ inline v16bf ld2(const __bf16* p0, const __bf16* p1) {
  V16U u;
  u.h[0] = *(const v8bf*)p0;
  u.h[1] = *(const v8bf*)p1;
  return u.v;
}

static __device__ inline v8bf cvt8(const float* p) {
  v8f f = *(const v8f*)p;
  v8bf r;
#pragma unroll
  for (int i = 0; i < 8; ++i) r[i] = (__bf16)f[i];
  return r;
}

static __device__ inline v8f wmma_bf16(v16bf a, v16bf b, v8f c) {
  // (neg_a, A, neg_b, B, c_mod, C, reuse_a, reuse_b)
  return __builtin_amdgcn_wmma_f32_16x16x32_bf16(false, a, false, b, (short)0, c,
                                                 false, false);
}

// ---------------------------------------------------------------------------
// Kernel 1: QKV projections, compile-time specialized per destination so the
// epilogue is straight-line stores (no per-element uniform branching).
// Z = 0 -> Q (scaled, both biases pre-added, two bf16 copies)
// Z = 1 -> K (bf16, row-major (h,n,dk))
// Z = 2 -> V (bf16, transposed (h,dv,n))
// Block tile 128 tokens x 64 cols, 8 waves each own a 16x64 strip (4 C tiles).
// LDS rows padded to 40 bf16 (80B) -> conflict-free 16-lane x 16B reads.
// ---------------------------------------------------------------------------
template <int Z>
__global__ __launch_bounds__(256) void k_qkv(
    const float* __restrict__ x, const float* __restrict__ W,
    const float* __restrict__ cb, const float* __restrict__ pb,
    __bf16* __restrict__ out0, __bf16* __restrict__ out1)
{
  __shared__ __bf16 xs[128 * 40];
  __shared__ __bf16 wsT[64 * 40];

  const int tid = threadIdx.x;
  const int lane = tid & 31, wv = tid >> 5;
  const int hf = lane >> 4, l16 = lane & 15;
  const int t0 = blockIdx.x * 128;
  const int c0 = blockIdx.y * 64;

  v8f acc[4] = {};

  for (int k0 = 0; k0 < DIMM; k0 += 32) {
    __syncthreads();
#pragma unroll
    for (int i = 0; i < 16; ++i) {              // 128x32 fp32 x-chunk -> bf16
      int e = tid + 256 * i, r = e >> 5, c = e & 31;
      xs[r * 40 + c] = (__bf16)x[(t0 + r) * DIMM + k0 + c];
    }
#pragma unroll
    for (int i = 0; i < 8; ++i) {               // 32x64 W-chunk, transposed
      int e = tid + 256 * i, r = e >> 6, c = e & 63;
      wsT[c * 40 + r] = (__bf16)W[(k0 + r) * 256 + c0 + c];
    }
    __syncthreads();

    const int lr = wv * 16 + l16;
    const int d0 = hf ? 16 : 0;
    v16bf a = ld2(&xs[lr * 40 + d0], &xs[lr * 40 + d0 + 8]);
#pragma unroll
    for (int ct = 0; ct < 4; ++ct) {
      int lc  = ct * 16 + l16;
      int kof = hf ? 16 : 0;
      v16bf b = ld2(&wsT[lc * 40 + kof], &wsT[lc * 40 + kof + 8]);
      acc[ct] = wmma_bf16(a, b, acc[ct]);
    }
  }

#pragma unroll
  for (int ct = 0; ct < 4; ++ct) {
    int col = c0 + ct * 16 + l16;
    int h = col >> 6, d = col & 63;
#pragma unroll
    for (int v = 0; v < 8; ++v) {
      int row = t0 + wv * 16 + v + 8 * hf;
      float val = acc[ct][v];
      if (Z == 0) {
        val *= QSCALE;
        out0[((size_t)h * NSEQ + row) * DK + d] = (__bf16)(val + cb[col]);
        out1[((size_t)h * NSEQ + row) * DK + d] = (__bf16)(val + pb[col]);
      } else if (Z == 1) {
        out0[((size_t)h * NSEQ + row) * DK + d] = (__bf16)val;
      } else {
        out0[((size_t)h * DK + d) * NSEQ + row] = (__bf16)val;  // transposed
      }
    }
  }
}

// ---------------------------------------------------------------------------
// Kernel 2: rel-pos keys. positions(8191,64) @ W_rel_k(64,256), features
// generated on the fly (central-mask + antisymmetric copy). 0.27 GF -> scalar.
// ---------------------------------------------------------------------------
__global__ __launch_bounds__(256) void k_relk(const float* __restrict__ Wr,
                                              __bf16* __restrict__ RelK)
{
  int idx = blockIdx.x * 256 + threadIdx.x;   // 8192 * 256 outputs
  int r = idx >> 8, col = idx & 255;
  int h = col >> 6, d = col & 63;
  float acc = 0.f;
  if (r < 2 * NSEQ - 1) {
    int dist = r - (NSEQ - 1);
    float ad  = fabsf((float)dist);
    float sgn = (dist > 0) ? 1.f : ((dist < 0) ? -1.f : 0.f);
    const float lw = __logf((float)(NSEQ + 1)) * (1.0f / 32.0f);
#pragma unroll 4
    for (int f = 0; f < 32; ++f) {
      float cw = __expf(lw * (float)(f + 1)) - 1.f;
      if (cw > ad)
        acc += Wr[f * 256 + col] + sgn * Wr[(f + 32) * 256 + col];
    }
  }
  RelK[((size_t)h * 8192 + r) * DK + d] = (__bf16)acc;
}

// ---------------------------------------------------------------------------
// Kernel 3: fused flash attention + Transformer-XL relative shift.
// One wave per 16 query rows; online softmax over keys in chunks of 32.
// rel logits: 16x48 band R over rel rows rbase..rbase+47, then ds_bpermute
// gather rel_logit[ii,jj] = R[ii, 15 + 16s + jj - ii].
// ---------------------------------------------------------------------------
__global__ __launch_bounds__(256) void k_attn(
    const __bf16* __restrict__ Qcb, const __bf16* __restrict__ Qpb,
    const __bf16* __restrict__ Kb, const __bf16* __restrict__ Vt,
    const __bf16* __restrict__ RelK, float* __restrict__ O)
{
  __shared__ __bf16 pbuf[8][16 * 32];  // per-wave P repack buffer (1 KB each)

  const int tid = threadIdx.x, lane = tid & 31, wv = tid >> 5;
  const int hf = lane >> 4, l16 = lane & 15;
  const int h  = blockIdx.y;
  const int i0 = blockIdx.x * 128 + wv * 16;

  const __bf16* q1 = Qcb + (size_t)h * NSEQ * DK;
  const __bf16* q2 = Qpb + (size_t)h * NSEQ * DK;
  const __bf16* kp = Kb  + (size_t)h * NSEQ * DK;
  const __bf16* vp = Vt  + (size_t)h * DK * NSEQ;
  const __bf16* rp = RelK + (size_t)h * 8192 * DK;
  float*        op = O   + (size_t)h * NSEQ * DK;

  // Q A-tiles (16x32 each, dk=64 -> two K-steps), resident for whole loop.
  v16bf aC[2], aP[2];
  {
    const __bf16* qr1 = q1 + (size_t)(i0 + l16) * DK;
    const __bf16* qr2 = q2 + (size_t)(i0 + l16) * DK;
    int d0 = hf ? 16 : 0;
    aC[0] = ld2(qr1 + d0,      qr1 + d0 + 8);
    aC[1] = ld2(qr1 + d0 + 32, qr1 + d0 + 40);
    aP[0] = ld2(qr2 + d0,      qr2 + d0 + 8);
    aP[1] = ld2(qr2 + d0 + 32, qr2 + d0 + 40);
  }

  v8f o[4] = {};
  float m[8], lsum[8];
#pragma unroll
  for (int v = 0; v < 8; ++v) { m[v] = -1e30f; lsum[v] = 0.f; }

  for (int jc = 0; jc < NSEQ; jc += 32) {
    // ---- content logits: two 16x16 tiles over 32 keys (4 WMMAs) ----
    v8f L[2];
#pragma unroll
    for (int s = 0; s < 2; ++s) {
      const __bf16* kr = kp + (size_t)(jc + 16 * s + l16) * DK;
      int d0 = hf ? 16 : 0;
      v16bf b0 = ld2(kr + d0,      kr + d0 + 8);
      v16bf b1 = ld2(kr + d0 + 32, kr + d0 + 40);
      v8f z = {};
      z = wmma_bf16(aC[0], b0, z);
      z = wmma_bf16(aC[1], b1, z);
      L[s] = z;
    }

    // ---- rel band R: 16x48 over rel rows rbase..rbase+47 (6 WMMAs) ----
    const int rbase = (NSEQ - 16) + jc - i0;   // 4080 + jc - i0, >= 0
    v8f R[3];
#pragma unroll
    for (int t = 0; t < 3; ++t) {
      int rr = rbase + 16 * t + l16;
      rr = (rr > 2 * NSEQ - 2) ? (2 * NSEQ - 2) : rr;   // clamp pad row
      const __bf16* rrp = rp + (size_t)rr * DK;
      int d0 = hf ? 16 : 0;
      v16bf b0 = ld2(rrp + d0,      rrp + d0 + 8);
      v16bf b1 = ld2(rrp + d0 + 32, rrp + d0 + 40);
      v8f z = {};
      z = wmma_bf16(aP[0], b0, z);
      z = wmma_bf16(aP[1], b1, z);
      R[t] = z;
    }

    // ---- diagonal gather (rel shift) via lane permutes ----
#pragma unroll
    for (int s = 0; s < 2; ++s) {
#pragma unroll
      for (int v = 0; v < 8; ++v) {
        int ii = v + 8 * hf;
        int c  = 15 + 16 * s + l16 - ii;          // 0..46
        int srcLane = (c & 15) | (hf << 4);
        float lo = __shfl(R[s][v],     srcLane, 32);
        float hi = __shfl(R[s + 1][v], srcLane, 32);
        L[s][v] += ((c >> 4) == s) ? lo : hi;
      }
    }

    // ---- online softmax (per-row stats live in 16-lane halves) ----
    float alpha[8];
#pragma unroll
    for (int v = 0; v < 8; ++v) {
      float cm = fmaxf(L[0][v], L[1][v]);
      cm = fmaxf(cm, __shfl_xor(cm, 1, 32));
      cm = fmaxf(cm, __shfl_xor(cm, 2, 32));
      cm = fmaxf(cm, __shfl_xor(cm, 4, 32));
      cm = fmaxf(cm, __shfl_xor(cm, 8, 32));
      float mn = fmaxf(m[v], cm);
      float al = __expf(m[v] - mn);
      float p0 = __expf(L[0][v] - mn);
      float p1 = __expf(L[1][v] - mn);
      float rs = p0 + p1;
      rs += __shfl_xor(rs, 1, 32);
      rs += __shfl_xor(rs, 2, 32);
      rs += __shfl_xor(rs, 4, 32);
      rs += __shfl_xor(rs, 8, 32);
      lsum[v] = lsum[v] * al + rs;
      m[v] = mn; alpha[v] = al;
      L[0][v] = p0; L[1][v] = p1;
    }
#pragma unroll
    for (int v = 0; v < 8; ++v) {
      o[0][v] *= alpha[v]; o[1][v] *= alpha[v];
      o[2][v] *= alpha[v]; o[3][v] *= alpha[v];
    }

    // ---- repack P (C-layout -> A-layout) through per-wave LDS ----
#pragma unroll
    for (int s = 0; s < 2; ++s)
#pragma unroll
      for (int v = 0; v < 8; ++v)
        pbuf[wv][(v + 8 * hf) * 32 + 16 * s + l16] = (__bf16)L[s][v];
    asm volatile("s_wait_dscnt 0" ::: "memory");

    {
      int pd0 = hf ? 16 : 0;
      v16bf aPk = ld2(&pbuf[wv][l16 * 32 + pd0], &pbuf[wv][l16 * 32 + pd0 + 8]);
#pragma unroll
      for (int dt = 0; dt < 4; ++dt) {
        const __bf16* vb = vp + (size_t)(dt * 16 + l16) * NSEQ + jc + (hf ? 16 : 0);
        v16bf bV = ld2(vb, vb + 8);
        o[dt] = wmma_bf16(aPk, bV, o[dt]);      // O += P @ V  (K = 32 keys)
      }
    }

    if (jc + 32 < NSEQ)                          // -> global_prefetch_b8
      __builtin_prefetch(kp + (size_t)(jc + 32 + lane) * DK, 0, 1);
  }

  // ---- finalize: O / l ----
#pragma unroll
  for (int v = 0; v < 8; ++v) {
    float inv = 1.0f / lsum[v];
    int row = i0 + v + 8 * hf;
#pragma unroll
    for (int dt = 0; dt < 4; ++dt)
      op[(size_t)row * DK + dt * 16 + l16] = o[dt][v] * inv;
  }
}

// ---------------------------------------------------------------------------
// Kernel 4: output projection. concat(O)(4096,256) @ Wo(256,1536) + bo.
// Final result stored non-temporal (write-once, never re-read on device).
// ---------------------------------------------------------------------------
__global__ __launch_bounds__(256) void k_out(
    const float* __restrict__ O, const float* __restrict__ Wo,
    const float* __restrict__ bo, float* __restrict__ out)
{
  __shared__ __bf16 wsT[64 * 40];

  const int tid = threadIdx.x, lane = tid & 31, wv = tid >> 5;
  const int hf = lane >> 4, l16 = lane & 15;
  const int t0 = blockIdx.x * 128;
  const int c0 = blockIdx.y * 64;
  const int arow = t0 + wv * 16 + l16;

  v8f acc[4] = {};

  for (int kk = 0; kk < 256; kk += 32) {
    __syncthreads();
#pragma unroll
    for (int i = 0; i < 8; ++i) {
      int e = tid + 256 * i, r = e >> 6, c = e & 63;
      wsT[c * 40 + r] = (__bf16)Wo[(kk + r) * DIMM + c0 + c];
    }
    __syncthreads();

    // A tile: feature index f = head-major concat; chunks stay inside a head.
    int f0 = kk + (hf ? 16 : 0), f1 = f0 + 8;
    V16U u;
    u.h[0] = cvt8(&O[(((size_t)(f0 >> 6)) * NSEQ + arow) * DK + (f0 & 63)]);
    u.h[1] = cvt8(&O[(((size_t)(f1 >> 6)) * NSEQ + arow) * DK + (f1 & 63)]);
    v16bf a = u.v;

#pragma unroll
    for (int ct = 0; ct < 4; ++ct) {
      int lc  = ct * 16 + l16;
      int kof = hf ? 16 : 0;
      v16bf b = ld2(&wsT[lc * 40 + kof], &wsT[lc * 40 + kof + 8]);
      acc[ct] = wmma_bf16(a, b, acc[ct]);
    }
  }

#pragma unroll
  for (int ct = 0; ct < 4; ++ct) {
    int col = c0 + ct * 16 + l16;
    float bias = bo[col];
#pragma unroll
    for (int v = 0; v < 8; ++v) {
      int r = t0 + wv * 16 + v + 8 * hf;
      __builtin_nontemporal_store(acc[ct][v] + bias,
                                  &out[(size_t)r * DIMM + col]);
    }
  }
}

// ---------------------------------------------------------------------------
extern "C" void kernel_launch(void* const* d_in, const int* in_sizes, int n_in,
                              void* d_out, int out_size, void* d_ws, size_t ws_size,
                              hipStream_t stream)
{
  const float* x  = (const float*)d_in[0];
  const float* Wq = (const float*)d_in[1];
  const float* Wk = (const float*)d_in[2];
  const float* Wv = (const float*)d_in[3];
  const float* Wr = (const float*)d_in[4];
  const float* Wo = (const float*)d_in[5];
  const float* bo = (const float*)d_in[6];
  const float* cb = (const float*)d_in[7];
  const float* pb = (const float*)d_in[8];
  float* out = (float*)d_out;

  char* ws = (char*)d_ws;
  __bf16* Qcb  = (__bf16*)(ws);
  __bf16* Qpb  = (__bf16*)(ws + (size_t)(2u << 20));
  __bf16* Kb   = (__bf16*)(ws + (size_t)(4u << 20));
  __bf16* Vt   = (__bf16*)(ws + (size_t)(6u << 20));
  __bf16* RelK = (__bf16*)(ws + (size_t)(8u << 20));
  float*  Oacc = (float*) (ws + (size_t)(12u << 20));

  dim3 gproj(NSEQ / 128, 4);
  k_qkv<0><<<gproj, 256, 0, stream>>>(x, Wq, cb, pb, Qcb, Qpb);
  k_qkv<1><<<gproj, 256, 0, stream>>>(x, Wk, cb, pb, Kb, nullptr);
  k_qkv<2><<<gproj, 256, 0, stream>>>(x, Wv, cb, pb, Vt, nullptr);
  k_relk<<<dim3(8192), 256, 0, stream>>>(Wr, RelK);
  k_attn<<<dim3(NSEQ / 128, NHEAD), 256, 0, stream>>>(Qcb, Qpb, Kb, Vt, RelK, Oacc);
  k_out <<<dim3(NSEQ / 128, DIMM / 64), 256, 0, stream>>>(Oacc, Wo, bo, out);
}
